// PaiNNLayer_25262997635094
// MI455X (gfx1250) — matrix operand order repaired and checked by reading, not compiled, not run on hardware
//
#include <hip/hip_runtime.h>
#include <math.h>

typedef __attribute__((ext_vector_type(16))) __bf16 v16bf;
typedef __attribute__((ext_vector_type(8)))  float  v8f;
typedef __attribute__((ext_vector_type(2)))  float  v2f;

#define FDIM 128
#define RDIM 20

__device__ __forceinline__ float silu_f(float x) { return x / (1.0f + __expf(-x)); }

__device__ __forceinline__ void atomic_add_f32(float* p, float v) {
  __hip_atomic_fetch_add(p, v, __ATOMIC_RELAXED, __HIP_MEMORY_SCOPE_AGENT);
}

// A operand: 16xK bf16 tile in LDS, row-major, K-chunk kc (32 wide).
// ISA 16-bit A layout: lanes 0-15 row M=lane hold K = kc*32 + {0..7, 16..23};
// lanes 16-31 hold K = kc*32 + {8..15, 24..31}.
__device__ __forceinline__ v16bf load_A_lds(const __bf16* a, int ldk, int kc, int lane) {
  const __bf16* p = a + (lane & 15) * ldk + kc * 32 + ((lane >> 4) << 3);
  v16bf A;
#pragma unroll
  for (int e = 0; e < 8; ++e) { A[e] = p[e]; A[8 + e] = p[16 + e]; }
  return A;
}

// B operand pre-packed in exact lane order: [nt*KC + kc][lane][16 bf16]
__device__ __forceinline__ v16bf load_B_packed(const __bf16* __restrict__ Wp, int KC,
                                               int nt, int kc, int lane) {
  return *(const v16bf*)(Wp + (((size_t)nt * KC + kc) * 32 + lane) * 16);
}

// ---------------- Weight pack: f32 (K,N) -> bf16 B-operand order (K mult of 32) ----------------
__global__ __launch_bounds__(256)
void pack_w_kernel(const float* __restrict__ W, __bf16* __restrict__ out, int K, int Nn) {
  int idx = blockIdx.x * 256 + threadIdx.x;
  if (idx >= K * Nn) return;
  int e     = idx & 15;
  int lane  = (idx >> 4) & 31;
  int chunk = idx >> 9;          // nt*KC + kc
  int KC = K >> 5;
  int kc = chunk % KC;
  int nt = chunk / KC;
  int k = kc * 32 + ((e >> 3) << 4) + ((lane >> 4) << 3) + (e & 7);
  int n = nt * 16 + (lane & 15);
  out[idx] = (__bf16)W[(size_t)k * Nn + n];
}

// ---------------- fW1 pack: f32 (20,128) -> f32 16x16x4 B-operand order ----------------
// layout [nt(8)][kc(5)][lane(32)][2 floats]; k = kc*4 + (lane>=16?2:0) + e, n = nt*16 + (lane&15)
__global__ __launch_bounds__(256)
void pack_w1_kernel(const float* __restrict__ W, float* __restrict__ out) {
  int idx = blockIdx.x * 256 + threadIdx.x;
  if (idx >= RDIM * FDIM) return;
  int e     = idx & 1;
  int lane  = (idx >> 1) & 31;
  int chunk = idx >> 6;          // nt*5 + kc
  int kc = chunk % 5;
  int nt = chunk / 5;
  int k = kc * 4 + ((lane >> 4) << 1) + e;
  int n = nt * 16 + (lane & 15);
  out[idx] = W[(size_t)k * FDIM + n];
}

// ---------------- Edge kernel: phi MLP + messages + scatter-add ----------------
__global__ __launch_bounds__(256)
void painn_edge_kernel(const float* __restrict__ s, const float* __restrict__ v,
                       const float* __restrict__ edge_attr, const float* __restrict__ rbf,
                       const float* __restrict__ fW1p, const float* __restrict__ fb1,
                       const __bf16* __restrict__ fW2p, const float* __restrict__ fb2,
                       const long long* __restrict__ edge_index,
                       float* __restrict__ m_s, float* __restrict__ m_v,
                       int N, int E) {
  __shared__ float  rbf_s[16][RDIM];
  __shared__ __bf16 h_s[16][FDIM];
  __shared__ float  phi_s[16][3 * FDIM];
  __shared__ float  dir_s[16][3];
  __shared__ int    row_s[16], col_s[16];

  const int tid  = threadIdx.x;
  const int lane = tid & 31;
  const int wave = tid >> 5;
  const int e0   = blockIdx.x * 16;

  for (int idx = tid; idx < 16 * RDIM; idx += 256) {
    int e = idx / RDIM, k = idx % RDIM;
    int ge = e0 + e;
    rbf_s[e][k] = (ge < E) ? rbf[(size_t)ge * RDIM + k] : 0.0f;
  }
  if (tid < 16) {
    int ge = e0 + tid;
    long long r = 0, c = 0;
    if (ge < E) { r = edge_index[ge]; c = edge_index[(size_t)E + ge]; }
    row_s[tid] = (int)r; col_s[tid] = (int)c;
  }
  if (tid >= 64 && tid < 112) {
    int idx = tid - 64; int e = idx / 3, d = idx % 3;
    int ge = e0 + e;
    dir_s[e][d] = (ge < E) ? edge_attr[(size_t)ge * 4 + 1 + d] : 0.0f;
  }
  __syncthreads();

  const int cn = lane & 15;
  const int rb = (lane >> 4) * 8;

  // h = silu(rbf @ fW1 + fb1) : M=16, K=20, N=128 via V_WMMA_F32_16X16X4_F32
  // (fp32 matrix pipe, no precision loss; one 16-wide N-tile per wave, 5 K-chunks)
  {
    int col = wave * 16 + cn;
    const int koff = (lane >> 4) << 1;
    v8f c = {};
#pragma unroll
    for (int kc = 0; kc < 5; ++kc) {
      v2f a = *(const v2f*)&rbf_s[lane & 15][kc * 4 + koff];
      v2f b = *(const v2f*)(fW1p + (((size_t)wave * 5 + kc) * 32 + lane) * 2);
      c = __builtin_amdgcn_wmma_f32_16x16x4_f32(false, a, false, b, (short)0, c, false, false);
    }
    float bias = fb1[col];
#pragma unroll
    for (int r = 0; r < 8; ++r) h_s[rb + r][col] = (__bf16)silu_f(c[r] + bias);
  }
  __syncthreads();

  // phi = h @ fW2 + fb2 : M=16, K=128, N=384 -> 24 N-tiles, 3 per wave.
  // A fragments loaded once, reused for all 3 tiles.
  v16bf a0 = load_A_lds(&h_s[0][0], FDIM, 0, lane);
  v16bf a1 = load_A_lds(&h_s[0][0], FDIM, 1, lane);
  v16bf a2 = load_A_lds(&h_s[0][0], FDIM, 2, lane);
  v16bf a3 = load_A_lds(&h_s[0][0], FDIM, 3, lane);
#pragma unroll
  for (int t = 0; t < 3; ++t) {
    int nt  = wave * 3 + t;
    int col = nt * 16 + cn;
    v8f c = {};
    c = __builtin_amdgcn_wmma_f32_16x16x32_bf16(false, a0, false, load_B_packed(fW2p, 4, nt, 0, lane), (short)0, c, false, false);
    c = __builtin_amdgcn_wmma_f32_16x16x32_bf16(false, a1, false, load_B_packed(fW2p, 4, nt, 1, lane), (short)0, c, false, false);
    c = __builtin_amdgcn_wmma_f32_16x16x32_bf16(false, a2, false, load_B_packed(fW2p, 4, nt, 2, lane), (short)0, c, false, false);
    c = __builtin_amdgcn_wmma_f32_16x16x32_bf16(false, a3, false, load_B_packed(fW2p, 4, nt, 3, lane), (short)0, c, false, false);
    float bias = fb2[col];
#pragma unroll
    for (int r = 0; r < 8; ++r) phi_s[rb + r][col] = c[r] + bias;
  }
  __syncthreads();

  // messages + segment-sum scatter (fp32 hardware atomics into L2-resident workspace)
#pragma unroll
  for (int i = 0; i < 8; ++i) {
    int idx = i * 256 + tid;
    int e = idx >> 7, f = idx & (FDIM - 1);
    int ge = e0 + e;
    if (ge < E) {
      int cc = col_s[e], rr = row_s[e];
      float sc  = s[(size_t)cc * FDIM + f];
      float pss = phi_s[e][f];
      float pvv = phi_s[e][FDIM + f];
      float psv = phi_s[e][2 * FDIM + f];
      atomic_add_f32(&m_s[(size_t)rr * FDIM + f], pss * sc);
#pragma unroll
      for (int d = 0; d < 3; ++d) {
        float mv = pvv * v[((size_t)cc * FDIM + f) * 3 + d] + psv * dir_s[e][d] * sc;
        atomic_add_f32(&m_v[((size_t)rr * FDIM + f) * 3 + d], mv);
      }
    }
  }
}

// ---------------- Node kernel: update MLP + output ----------------
__global__ __launch_bounds__(256)
void painn_node_kernel(const float* __restrict__ s, const float* __restrict__ v,
                       const __bf16* __restrict__ uW1p, const float* __restrict__ ub1,
                       const __bf16* __restrict__ uW2p, const float* __restrict__ ub2,
                       const float* __restrict__ m_s, const float* __restrict__ m_v,
                       float* __restrict__ out_s, float* __restrict__ out_v, int N) {
  __shared__ __bf16 x_s[16][3 * FDIM];
  __shared__ __bf16 h_s[16][FDIM];
  __shared__ float  upd_s[16][3 * FDIM];

  const int tid  = threadIdx.x;
  const int lane = tid & 31;
  const int wave = tid >> 5;
  const int n0   = blockIdx.x * 16;

  // x = concat[s, m_s, ||m_v||]  (16 x 384, bf16)
  for (int idx = tid; idx < 16 * 3 * FDIM; idx += 256) {
    int i = idx / (3 * FDIM), c = idx % (3 * FDIM);
    int gn = n0 + i;
    float val = 0.0f;
    if (gn < N) {
      if (c < FDIM)            val = s[(size_t)gn * FDIM + c];
      else if (c < 2 * FDIM)   val = m_s[(size_t)gn * FDIM + (c - FDIM)];
      else {
        const float* p = &m_v[((size_t)gn * FDIM + (c - 2 * FDIM)) * 3];
        val = sqrtf(p[0] * p[0] + p[1] * p[1] + p[2] * p[2]);
      }
    }
    x_s[i][c] = (__bf16)val;
  }
  __syncthreads();

  const int cn = lane & 15;
  const int rb = (lane >> 4) * 8;

  // h = silu(x @ uW1 + ub1) : M=16, K=384, N=128 -> one N-tile per wave
  {
    int col = wave * 16 + cn;
    v8f c = {};
#pragma unroll
    for (int kc = 0; kc < 12; ++kc) {
      v16bf a = load_A_lds(&x_s[0][0], 3 * FDIM, kc, lane);
      v16bf b = load_B_packed(uW1p, 12, wave, kc, lane);
      c = __builtin_amdgcn_wmma_f32_16x16x32_bf16(false, a, false, b, (short)0, c, false, false);
    }
    float bias = ub1[col];
#pragma unroll
    for (int r = 0; r < 8; ++r) h_s[rb + r][col] = (__bf16)silu_f(c[r] + bias);
  }
  __syncthreads();

  // upd = h @ uW2 + ub2 : M=16, K=128, N=384 -> 3 N-tiles per wave, A reused
  v16bf a0 = load_A_lds(&h_s[0][0], FDIM, 0, lane);
  v16bf a1 = load_A_lds(&h_s[0][0], FDIM, 1, lane);
  v16bf a2 = load_A_lds(&h_s[0][0], FDIM, 2, lane);
  v16bf a3 = load_A_lds(&h_s[0][0], FDIM, 3, lane);
#pragma unroll
  for (int t = 0; t < 3; ++t) {
    int nt  = wave * 3 + t;
    int col = nt * 16 + cn;
    v8f c = {};
    c = __builtin_amdgcn_wmma_f32_16x16x32_bf16(false, a0, false, load_B_packed(uW2p, 4, nt, 0, lane), (short)0, c, false, false);
    c = __builtin_amdgcn_wmma_f32_16x16x32_bf16(false, a1, false, load_B_packed(uW2p, 4, nt, 1, lane), (short)0, c, false, false);
    c = __builtin_amdgcn_wmma_f32_16x16x32_bf16(false, a2, false, load_B_packed(uW2p, 4, nt, 2, lane), (short)0, c, false, false);
    c = __builtin_amdgcn_wmma_f32_16x16x32_bf16(false, a3, false, load_B_packed(uW2p, 4, nt, 3, lane), (short)0, c, false, false);
    float bias = ub2[col];
#pragma unroll
    for (int r = 0; r < 8; ++r) upd_s[rb + r][col] = c[r] + bias;
  }
  __syncthreads();

  // s_out = s + delta_s ; v_out = alpha*v + beta*m_v
#pragma unroll
  for (int i = 0; i < 8; ++i) {
    int idx = i * 256 + tid;
    int nl = idx >> 7, f = idx & (FDIM - 1);
    int gn = n0 + nl;
    if (gn < N) {
      float ds    = upd_s[nl][f];
      float alpha = upd_s[nl][FDIM + f];
      float beta  = upd_s[nl][2 * FDIM + f];
      size_t si = (size_t)gn * FDIM + f;
      out_s[si] = s[si] + ds;
#pragma unroll
      for (int d = 0; d < 3; ++d)
        out_v[si * 3 + d] = alpha * v[si * 3 + d] + beta * m_v[si * 3 + d];
    }
  }
}

extern "C" void kernel_launch(void* const* d_in, const int* in_sizes, int n_in,
                              void* d_out, int out_size, void* d_ws, size_t ws_size,
                              hipStream_t stream) {
  const float* s         = (const float*)d_in[0];
  const float* v         = (const float*)d_in[1];
  const float* edge_attr = (const float*)d_in[2];
  const float* rbf       = (const float*)d_in[3];
  const float* fW1       = (const float*)d_in[4];
  const float* fb1       = (const float*)d_in[5];
  const float* fW2       = (const float*)d_in[6];
  const float* fb2       = (const float*)d_in[7];
  const float* uW1       = (const float*)d_in[8];
  const float* ub1       = (const float*)d_in[9];
  const float* uW2       = (const float*)d_in[10];
  const float* ub2       = (const float*)d_in[11];
  const long long* edge_index = (const long long*)d_in[12];

  const int N = in_sizes[0] / FDIM;   // s is (N, F)
  const int E = in_sizes[2] / 4;      // edge_attr is (E, 4)

  float*  m_s  = (float*)d_ws;                             // N*F floats
  float*  m_v  = m_s + (size_t)N * FDIM;                   // N*F*3 floats
  float*  fW1p = m_v + (size_t)N * FDIM * 3;               // 20*128 f32 (packed)
  __bf16* fW2p = (__bf16*)(fW1p + RDIM * FDIM);            // 128*384 bf16
  __bf16* uW1p = fW2p + FDIM * 3 * FDIM;                   // 384*128 bf16
  __bf16* uW2p = uW1p + 3 * FDIM * FDIM;                   // 128*384 bf16

  // zero segment-sum accumulators each call (graph-capture legal)
  hipMemsetAsync(d_ws, 0, (size_t)N * FDIM * 4 * sizeof(float), stream);

  // pack weights into WMMA B-operand lane order (cheap, once per call)
  pack_w1_kernel<<<(RDIM * FDIM + 255) / 256, 256, 0, stream>>>(fW1, fW1p);
  pack_w_kernel<<<(FDIM * 3 * FDIM + 255) / 256, 256, 0, stream>>>(fW2, fW2p, FDIM, 3 * FDIM);
  pack_w_kernel<<<(3 * FDIM * FDIM + 255) / 256, 256, 0, stream>>>(uW1, uW1p, 3 * FDIM, FDIM);
  pack_w_kernel<<<(FDIM * 3 * FDIM + 255) / 256, 256, 0, stream>>>(uW2, uW2p, FDIM, 3 * FDIM);

  int eblocks = (E + 15) / 16;
  painn_edge_kernel<<<eblocks, 256, 0, stream>>>(s, v, edge_attr, rbf, fW1p, fb1, fW2p, fb2,
                                                 edge_index, m_s, m_v, N, E);

  float* out_s = (float*)d_out;
  float* out_v = out_s + (size_t)N * FDIM;
  int nblocks = (N + 15) / 16;
  painn_node_kernel<<<nblocks, 256, 0, stream>>>(s, v, uW1p, ub1, uW2p, ub2, m_s, m_v,
                                                 out_s, out_v, N);
}
